// GATNetReduced_26620207301226
// MI455X (gfx1250) — matrix operand order repaired and checked by reading, not compiled, not run on hardware
//
#include <hip/hip_runtime.h>

typedef __attribute__((ext_vector_type(16))) _Float16 v16h;
typedef __attribute__((ext_vector_type(8)))  _Float16 v8h;
typedef __attribute__((ext_vector_type(8)))  float    v8f;
typedef __attribute__((ext_vector_type(4)))  float    v4f;

#define DIN   512
#define DCAT  1024
#define DHEAD 512
#define NHEAD 2
#define NEG_SLOPE 0.2f

union AFrag { v16h v; v8h h[2]; };

// ---------------------------------------------------------------------------
// f32 -> f16 elementwise convert
__global__ void conv_f16_kernel(const float* __restrict__ src,
                                _Float16* __restrict__ dst, int n) {
  int i = blockIdx.x * blockDim.x + threadIdx.x;
  if (i < n) dst[i] = (_Float16)src[i];
}

// f32 [rows,cols] -> f16 [cols,rows]  (transpose + convert; weights only, tiny)
__global__ void transpose_f16_kernel(const float* __restrict__ src,
                                     _Float16* __restrict__ dst,
                                     int rows, int cols) {
  int i = blockIdx.x * blockDim.x + threadIdx.x;
  if (i >= rows * cols) return;
  int r = i / cols, c = i % cols;
  dst[(size_t)c * rows + r] = (_Float16)src[i];
}

__global__ void fill_i32_kernel(int* __restrict__ p, int v, int n) {
  int i = blockIdx.x * blockDim.x + threadIdx.x;
  if (i < n) p[i] = v;
}

// ---------------------------------------------------------------------------
// Fragment loaders (contiguous k order: lane half h holds k = 16h+[0..15], wave32)
__device__ __forceinline__ void load_frags(const _Float16* __restrict__ arow,
                                           const _Float16* __restrict__ brow,
                                           int K, int k, AFrag& a, AFrag* b) {
  a.h[0] = *(const v8h*)(arow + k);
  a.h[1] = *(const v8h*)(arow + k + 8);
#pragma unroll
  for (int s = 0; s < 4; ++s) {
    const _Float16* bp = brow + (size_t)(s * 16) * K + k;
    b[s].h[0] = *(const v8h*)(bp);
    b[s].h[1] = *(const v8h*)(bp + 8);
  }
}

__device__ __forceinline__ void wmma4(v8f* acc, const AFrag& a, const AFrag* b) {
#pragma unroll
  for (int s = 0; s < 4; ++s)
    acc[s] = __builtin_amdgcn_wmma_f32_16x16x32_f16(
        false, a.v, false, b[s].v, (short)0, acc[s], false, false);
}

// Generic WMMA GEMM:  C[M,Nc] = A[M,K](f16,row) * Bt[Nc,K](f16,row)^T
// One wave computes a 16x64 strip (1 A fragment reused by 4 WMMAs per K-step).
// K-loop is software-pipelined with a 2-deep register double buffer so global
// loads for step k+32 are outstanding while WMMAs for step k execute.
// Requires K % 64 == 0 (holds: 512, 1024, 128).
__global__ void wmma_gemm_kernel(const _Float16* __restrict__ A,
                                 const _Float16* __restrict__ Bt,
                                 const float* __restrict__ bias,
                                 float* __restrict__ Cf,
                                 _Float16* __restrict__ Ch,
                                 int M, int K, int Nc, int relu) {
  int wid  = (blockIdx.x * blockDim.x + threadIdx.x) >> 5;
  int lane = threadIdx.x & 31;
  int tilesN = Nc >> 6;                 // 64-wide strips
  int tm = wid / tilesN;
  int tn = (wid % tilesN) << 6;
  if (tm * 16 >= M) return;

  int half = lane >> 4;                 // which half-wave
  int mrow = lane & 15;                 // A row / B col within tile

  // A: lanes 0-15 hold K = k0+[0..15]; lanes 16-31 K = k0+[16..31]
  const _Float16* arow = A  + (size_t)(tm * 16 + mrow) * K + (half << 4);
  // B (transposed): lanes 0-15 hold K = k0+[0..15]; lanes 16-31 K = k0+[16..31]
  const _Float16* brow = Bt + (size_t)(tn + mrow) * K + (half << 4);

  v8f acc[4];
#pragma unroll
  for (int s = 0; s < 4; ++s) acc[s] = (v8f){0.f,0.f,0.f,0.f,0.f,0.f,0.f,0.f};

  AFrag a0, b0[4], a1, b1[4];
  load_frags(arow, brow, K, 0, a0, b0);
  for (int k0 = 0; k0 < K; k0 += 64) {
    load_frags(arow, brow, K, k0 + 32, a1, b1);   // prefetch odd step
    wmma4(acc, a0, b0);
    int k2 = k0 + 64;
    if (k2 < K) load_frags(arow, brow, K, k2, a0, b0);  // prefetch next even
    wmma4(acc, a1, b1);
  }

  // C layout: VGPR j -> row = tm*16 + j + 8*half ; col = tn + s*16 + mrow
  int rbase = tm * 16 + (half << 3);
#pragma unroll
  for (int s = 0; s < 4; ++s) {
    int col = tn + s * 16 + mrow;
    float bv = bias ? bias[col] : 0.f;
#pragma unroll
    for (int j = 0; j < 8; ++j) {
      float v = acc[s][j] + bv;
      if (relu) v = fmaxf(v, 0.f);
      size_t idx = (size_t)(rbase + j) * Nc + col;
      if (Cf) Cf[idx] = v;
      if (Ch) Ch[idx] = (_Float16)v;
    }
  }
}

// ---------------------------------------------------------------------------
// a_src[n,h] = dot(xh[n,h,:], att_src[h,:]) ; same for a_dst. One wave per (n,h).
__global__ void att_reduce_kernel(const float* __restrict__ xh,
                                  const float* __restrict__ att_src,
                                  const float* __restrict__ att_dst,
                                  float* __restrict__ a_src,
                                  float* __restrict__ a_dst, int N) {
  int wid  = (blockIdx.x * blockDim.x + threadIdx.x) >> 5;
  int lane = threadIdx.x & 31;
  if (wid >= N * NHEAD) return;
  int n = wid >> 1, h = wid & 1;
  const float* row = xh + (size_t)n * DCAT + h * DHEAD;
  const float* as  = att_src + h * DHEAD;
  const float* ad  = att_dst + h * DHEAD;
  float s0 = 0.f, s1 = 0.f;
  for (int k = lane; k < DHEAD; k += 32) {
    float v = row[k];
    s0 += v * as[k];
    s1 += v * ad[k];
  }
#pragma unroll
  for (int off = 16; off > 0; off >>= 1) {
    s0 += __shfl_down(s0, off, 32);
    s1 += __shfl_down(s1, off, 32);
  }
  if (lane == 0) { a_src[wid] = s0; a_dst[wid] = s1; }
}

__device__ __forceinline__ int f32_order_key(float f) {
  int i = __float_as_int(f);
  return i >= 0 ? i : (i ^ 0x7fffffff);
}
__device__ __forceinline__ float f32_order_inv(int k) {
  return __int_as_float(k >= 0 ? k : (k ^ 0x7fffffff));
}

// edge pass 1: leaky-relu logits + segment max over dst (ordered-int atomicMax)
__global__ void edge_logit_max_kernel(const int* __restrict__ ei,
                                      const float* __restrict__ a_src,
                                      const float* __restrict__ a_dst,
                                      float* __restrict__ logits,
                                      int* __restrict__ m_i,
                                      int E, int Etot) {
  int e = blockIdx.x * blockDim.x + threadIdx.x;
  if (e >= Etot) return;
  int s, d;
  if (e < E) { s = ei[e]; d = ei[E + e]; } else { s = d = e - E; }
#pragma unroll
  for (int h = 0; h < NHEAD; ++h) {
    float l = a_src[s * NHEAD + h] + a_dst[d * NHEAD + h];
    l = l > 0.f ? l : NEG_SLOPE * l;
    logits[(size_t)e * NHEAD + h] = l;
    atomicMax(&m_i[d * NHEAD + h], f32_order_key(l));
  }
}

// edge pass 2: e = exp(l - m[dst]); z[dst] += e (overwrite logits with e)
__global__ void edge_exp_sum_kernel(const int* __restrict__ ei,
                                    float* __restrict__ logits,
                                    const int* __restrict__ m_i,
                                    float* __restrict__ z,
                                    int E, int Etot) {
  int e = blockIdx.x * blockDim.x + threadIdx.x;
  if (e >= Etot) return;
  int d = (e < E) ? ei[E + e] : (e - E);
#pragma unroll
  for (int h = 0; h < NHEAD; ++h) {
    float l  = logits[(size_t)e * NHEAD + h];
    float mv = f32_order_inv(m_i[d * NHEAD + h]);
    float ev = __expf(l - mv);
    logits[(size_t)e * NHEAD + h] = ev;
    unsafeAtomicAdd(&z[d * NHEAD + h], ev);
  }
}

// edge pass 3: out[dst] += alpha * xh[src] ; one block (256 thr) per edge,
// each thread scatters 4 contiguous channels with f32 atomics.
__global__ void edge_scatter_kernel(const int* __restrict__ ei,
                                    const float* __restrict__ logits,
                                    const float* __restrict__ z,
                                    const float* __restrict__ xh,
                                    float* __restrict__ out,
                                    int E, int Etot) {
  int e = blockIdx.x;
  if (e >= Etot) return;
  int s, d;
  if (e < E) { s = ei[e]; d = ei[E + e]; } else { s = d = e - E; }
  int c = threadIdx.x * 4;            // 0..1023
  int h = c >> 9;                     // head for this channel group
  float coef = logits[(size_t)e * NHEAD + h] /
               fmaxf(z[d * NHEAD + h], 1e-16f);
  v4f xv = *(const v4f*)(xh + (size_t)s * DCAT + c);
  float* op = out + (size_t)d * DCAT + c;
  unsafeAtomicAdd(op + 0, coef * xv.x);
  unsafeAtomicAdd(op + 1, coef * xv.y);
  unsafeAtomicAdd(op + 2, coef * xv.z);
  unsafeAtomicAdd(op + 3, coef * xv.w);
}

// h = relu(agg + bias) -> f16
__global__ void bias_relu_f16_kernel(const float* __restrict__ agg,
                                     const float* __restrict__ bias,
                                     _Float16* __restrict__ hh, int n) {
  int i = blockIdx.x * blockDim.x + threadIdx.x;
  if (i >= n) return;
  float v = agg[i] + bias[i & (DCAT - 1)];
  hh[i] = (_Float16)fmaxf(v, 0.f);
}

// h3 = h64 @ W2 + b2 (64x3, tiny); pack {x,y,z,||h||^2} as one 16B record per
// node so cdist can pull it with one async B128 per entry.
__global__ void head_mlp_kernel(const float* __restrict__ h64,
                                const float* __restrict__ W2,
                                const float* __restrict__ b2,
                                v4f* __restrict__ h3p, int N) {
  int n = blockIdx.x * blockDim.x + threadIdx.x;
  if (n >= N) return;
  const float* r = h64 + (size_t)n * 64;
  float o[3];
#pragma unroll
  for (int j = 0; j < 3; ++j) {
    float acc = b2[j];
    for (int k = 0; k < 64; ++k) acc += r[k] * W2[k * 3 + j];
    o[j] = acc;
  }
  v4f rec;
  rec.x = o[0]; rec.y = o[1]; rec.z = o[2];
  rec.w = o[0] * o[0] + o[1] * o[1] + o[2] * o[2];
  h3p[n] = rec;
}

// cdist: out[i,j] = sqrt(max(sq[i]+sq[j]-2*dot,0)).
// j-slab of packed records is staged straight into LDS with CDNA5 async
// global->LDS B128 copies (ASYNCcnt), then each thread emits 4 outputs with
// nontemporal stores (1 GB streaming output).
__global__ void cdist_kernel(const v4f* __restrict__ h3p,
                             float* __restrict__ out, int N) {
  __shared__ v4f tile[1024];
  int jbase = blockIdx.x << 10;
  int i = blockIdx.y;
#pragma unroll
  for (int u = 0; u < 4; ++u) {
    int j = threadIdx.x + u * 256;
    const v4f* gp = h3p + jbase + j;
    unsigned ldsoff = (unsigned)(uintptr_t)&tile[j];  // low 32b of flat LDS addr
    asm volatile("global_load_async_to_lds_b128 %0, %1, off"
                 :: "v"(ldsoff), "v"(gp) : "memory");
  }
  asm volatile("s_wait_asynccnt 0x0" ::: "memory");
  __syncthreads();

  v4f me = h3p[i];
  int j0 = threadIdx.x * 4;
  v4f r;
#pragma unroll
  for (int u = 0; u < 4; ++u) {
    v4f t = tile[j0 + u];
    float dot = me.x * t.x + me.y * t.y + me.z * t.z;
    float d2 = me.w + t.w - 2.f * dot;
    d2 = fmaxf(d2, 0.f);
    r[u] = d2 > 0.f ? sqrtf(d2) : 0.f;
  }
  __builtin_nontemporal_store(r, (v4f*)(out + (size_t)i * N + jbase + j0));
}

// ---------------------------------------------------------------------------
extern "C" void kernel_launch(void* const* d_in, const int* in_sizes, int n_in,
                              void* d_out, int out_size, void* d_ws, size_t ws_size,
                              hipStream_t stream) {
  const float* x      = (const float*)d_in[0];
  const int*   ei     = (const int*)d_in[1];
  const float* W      = (const float*)d_in[2];
  const float* attS   = (const float*)d_in[3];
  const float* attD   = (const float*)d_in[4];
  const float* bias   = (const float*)d_in[5];
  const float* Wa     = (const float*)d_in[6];
  const float* ba     = (const float*)d_in[7];
  const float* W1     = (const float*)d_in[8];
  const float* b1     = (const float*)d_in[9];
  const float* W2     = (const float*)d_in[10];
  const float* b2     = (const float*)d_in[11];
  float* outp = (float*)d_out;

  const int N    = in_sizes[0] / DIN;        // 16384
  const int E    = in_sizes[1] / 2;          // 524288
  const int Etot = E + N;

  // --- big intermediates live inside d_out (1 GB; only cdist writes it last)
  char* ob = (char*)d_out;
  float*    xh     = (float*)(ob);                               // 64 MB
  float*    agg    = (float*)(ob + (size_t)N * DCAT * 4);        // 64 MB
  _Float16* x_h    = (_Float16*)(ob + (size_t)2 * N * DCAT * 4); // 16 MB
  _Float16* h_h    = (_Float16*)xh;        // aliases xh after it's dead (32 MB)
  _Float16* h128_h = x_h;                  // aliases x_h after GEMM1     (4 MB)
  float*    h64    = agg;                  // aliases agg after bias_relu (4 MB)

  // --- small buffers in workspace (~7 MB)
  char* wb = (char*)d_ws;
  size_t o = 0;
  auto carve = [&](size_t bytes) { void* p = wb + o; o += (bytes + 255) & ~(size_t)255; return p; };
  _Float16* Wt     = (_Float16*)carve((size_t)DIN * DCAT * 2);   // [1024,512]
  _Float16* Wat    = (_Float16*)carve((size_t)DCAT * 128 * 2);   // [128,1024]
  _Float16* W1t    = (_Float16*)carve((size_t)128 * 64 * 2);     // [64,128]
  float*    a_src  = (float*)carve((size_t)N * NHEAD * 4);
  float*    a_dst  = (float*)carve((size_t)N * NHEAD * 4);
  int*      m_i    = (int*)carve((size_t)N * NHEAD * 4);
  float*    zbuf   = (float*)carve((size_t)N * NHEAD * 4);
  float*    logits = (float*)carve((size_t)Etot * NHEAD * 4);
  v4f*      h3p    = (v4f*)carve((size_t)N * 16);
  (void)ws_size; (void)n_in; (void)out_size;

  const int T = 256;
  auto blocks = [](long long n, int t) { return (unsigned)((n + t - 1) / t); };

  // 1) convert/transpose to f16
  conv_f16_kernel<<<blocks((long long)N * DIN, T), T, 0, stream>>>(x, x_h, N * DIN);
  transpose_f16_kernel<<<blocks((long long)DIN * DCAT, T), T, 0, stream>>>(W, Wt, DIN, DCAT);
  transpose_f16_kernel<<<blocks((long long)DCAT * 128, T), T, 0, stream>>>(Wa, Wat, DCAT, 128);
  transpose_f16_kernel<<<blocks((long long)128 * 64, T), T, 0, stream>>>(W1, W1t, 128, 64);

  // 2) xh = x @ W   (WMMA f16->f32)
  {
    long long waves = (long long)(N / 16) * (DCAT / 64);
    wmma_gemm_kernel<<<blocks(waves * 32, T), T, 0, stream>>>(
        x_h, Wt, nullptr, xh, nullptr, N, DIN, DCAT, 0);
  }

  // 3) attention coefficients
  att_reduce_kernel<<<blocks((long long)N * NHEAD * 32, T), T, 0, stream>>>(
      xh, attS, attD, a_src, a_dst, N);

  // 4) init segment stats + aggregation buffer
  fill_i32_kernel<<<blocks(N * NHEAD, T), T, 0, stream>>>(m_i, (int)0x80000000, N * NHEAD);
  hipMemsetAsync(zbuf, 0, (size_t)N * NHEAD * 4, stream);
  hipMemsetAsync(agg, 0, (size_t)N * DCAT * 4, stream);

  // 5) segment softmax + scatter
  edge_logit_max_kernel<<<blocks(Etot, T), T, 0, stream>>>(ei, a_src, a_dst, logits, m_i, E, Etot);
  edge_exp_sum_kernel<<<blocks(Etot, T), T, 0, stream>>>(ei, logits, m_i, zbuf, E, Etot);
  edge_scatter_kernel<<<Etot, T, 0, stream>>>(ei, logits, zbuf, xh, agg, E, Etot);

  // 6) h = relu(agg + bias) -> f16 (aliases xh region)
  bias_relu_f16_kernel<<<blocks((long long)N * DCAT, T), T, 0, stream>>>(agg, bias, h_h, N * DCAT);

  // 7) h128 = relu(h @ Wa + ba)  (WMMA, f16 out)
  {
    long long waves = (long long)(N / 16) * (128 / 64);
    wmma_gemm_kernel<<<blocks(waves * 32, T), T, 0, stream>>>(
        h_h, Wat, ba, nullptr, h128_h, N, DCAT, 128, 1);
  }

  // 8) h64 = relu(h128 @ W1 + b1)  (WMMA, f32 out; aliases agg region)
  {
    long long waves = (long long)(N / 16) * (64 / 64);
    wmma_gemm_kernel<<<blocks(waves * 32, T), T, 0, stream>>>(
        h128_h, W1t, b1, h64, nullptr, N, 128, 64, 1);
  }

  // 9) h3/sq packed records
  head_mlp_kernel<<<blocks(N, T), T, 0, stream>>>(h64, W2, b2, h3p, N);

  // 10) cdist -> d_out (async LDS staging; overwrites all scratch in d_out)
  {
    dim3 g(N / 1024, N);
    cdist_kernel<<<g, T, 0, stream>>>(h3p, outp, N);
  }
}
